// actor_ReNN_19164144075353
// MI455X (gfx1250) — matrix operand order repaired and checked by reading, not compiled, not run on hardware
//
#include <hip/hip_runtime.h>

typedef __attribute__((ext_vector_type(16))) _Float16 v16h;
typedef __attribute__((ext_vector_type(8)))  float    v8f;

#define DD    64
#define NOBJ_ 16
#define NBLK  3
#define OBSW  298
#define FSTR  65      // f32 LDS row stride (floats)
#define ASTR  68      // f16 LDS row stride (halfs, even -> dword aligned rows)

__device__ __forceinline__ float lrelu(float x) { return x > 0.f ? x : 0.01f * x; }

// Native CDNA5 transcendental tanh (V_TANH_F32). Builtin path lets the compiler
// manage the TRANS-op hazard; asm fallback inserts the required 1-op gap itself.
__device__ __forceinline__ float fast_tanh(float x) {
#if __has_builtin(__builtin_amdgcn_tanhf)
  return __builtin_amdgcn_tanhf(x);
#else
  float r;
  asm volatile("v_tanh_f32 %0, %1\n\tv_nop" : "=v"(r) : "v"(x));
  return r;
#endif
}

__device__ __forceinline__ v8f wmma16(v16h a, v16h b, v8f c) {
  return __builtin_amdgcn_wmma_f32_16x16x32_f16(false, a, false, b, (short)0, c, false, false);
}

// Gather a 16x32 f16 A-fragment (ISA 7.12.2 layout) from a row-major LDS tile.
__device__ __forceinline__ v16h load_afrag(const _Float16* tile, int kt, int lane) {
  union { v16h h; unsigned u[8]; } A;
  const _Float16* row = tile + (lane & 15) * ASTR + kt * 32 + ((lane >> 4) ? 8 : 0);
#pragma unroll
  for (int j = 0; j < 8; ++j) {
    int k2 = 2 * j;
    int k  = k2 + (k2 >= 8 ? 8 : 0);            // pairs: 0,2,4,6,16,18,20,22
    A.u[j] = *(const unsigned*)(row + k);        // dword LDS load (two halfs)
  }
  return A.h;
}

// ---------------- prep kernels (run each launch; deterministic) ----------------

// WBIG[i] (64x192 f32) = [ Wq_i @ Wch_i | Wc_i | Wm_i ]
__global__ void prep_fuse(const float* __restrict__ Wqcm, const float* __restrict__ Wch,
                          float* __restrict__ WBIG) {
  int ik = blockIdx.x; int i = ik >> 6; int k = ik & 63;
  int n = threadIdx.x;                       // 0..191
  const float* wq = Wqcm + (size_t)i * 64 * 192;
  float v;
  if (n < 64) {
    const float* wc = Wch + (size_t)i * 64 * 64;
    float s = 0.f;
    for (int t = 0; t < 64; ++t) s += wq[k * 192 + t] * wc[t * 64 + n];
    v = s;
  } else {
    v = wq[k * 192 + n];
  }
  WBIG[(size_t)i * 64 * 192 + k * 192 + n] = v;
}

// BBIG[i] (192) = [ bq@Wch+bch | bc | bm ]; QHP (64) = pq@pWch + pbch
__global__ void prep_bias(const float* __restrict__ bqcm, const float* __restrict__ Wch,
                          const float* __restrict__ bch,  const float* __restrict__ pq,
                          const float* __restrict__ pWch, const float* __restrict__ pbch,
                          float* __restrict__ bbig, float* __restrict__ qhp) {
  int i = blockIdx.x; int n = threadIdx.x;
  if (i < 3) {
    float v;
    if (n < 64) {
      float s = bch[i * 64 + n];
      for (int t = 0; t < 64; ++t) s += bqcm[i * 192 + t] * Wch[(size_t)i * 4096 + t * 64 + n];
      v = s;
    } else v = bqcm[i * 192 + n];
    bbig[i * 192 + n] = v;
  } else if (n < 64) {
    float s = pbch[n];
    for (int t = 0; t < 64; ++t) s += pq[t] * pWch[t * 64 + n];
    qhp[n] = s;
  }
}

// Pack a KxN f32 weight into f16 WMMA B-fragments (32x16 tiles, tile = nt*KT+kt).
__global__ void pack_frags(const float* __restrict__ src, int Kvalid, int ld,
                           int NT, int KT, _Float16* __restrict__ dst) {
  int tid = blockIdx.x * blockDim.x + threadIdx.x;
  int total = NT * KT * 32;
  if (tid >= total) return;
  int lane = tid & 31; int tileIdx = tid >> 5;
  int nt = tileIdx / KT, kt = tileIdx % KT;
  int n  = nt * 16 + (lane & 15);
  int k0 = kt * 32 + ((lane >= 16) ? 8 : 0);
  _Float16* d = dst + (size_t)tid * 16;
  for (int e = 0; e < 16; ++e) {
    int k = k0 + e + (e >= 8 ? 8 : 0);
    d[e] = (_Float16)((k < Kvalid) ? src[(size_t)k * ld + n] : 0.f);
  }
}

// ---------------- fused actor kernel: one batch element per wave ----------------

__global__ __launch_bounds__(64) void actor_main(
    const float* __restrict__ obs, const float* __restrict__ b_in,
    const float* __restrict__ Wlg, const float* __restrict__ blg,
    const float* __restrict__ temp, const float* __restrict__ brh,
    const float* __restrict__ pWlg, const float* __restrict__ pblg,
    const float* __restrict__ pWrh, const float* __restrict__ pbrh,
    const float* __restrict__ ptemp,
    const float* __restrict__ Wo1, const float* __restrict__ bo1,
    const float* __restrict__ Wo2, const float* __restrict__ bo2,
    const float* __restrict__ Wo3, const float* __restrict__ bo3,
    const float* __restrict__ Wo4, const float* __restrict__ bo4,
    const _Float16* __restrict__ fragWin, const _Float16* __restrict__ fragWbig,
    const _Float16* __restrict__ fragWrh,
    const float* __restrict__ bbig, const float* __restrict__ qhp,
    float* __restrict__ out, int B)
{
  __shared__ __align__(16) _Float16 sh_act[2][16 * ASTR];
  __shared__ __align__(16) float sh_qh[2][16 * FSTR];
  __shared__ __align__(16) float sh_c [2][16 * FSTR];
  __shared__ __align__(16) float sh_m [2][16 * FSTR];
  __shared__ __align__(16) float sh_p [2][16 * 17];
  __shared__ __align__(16) float sh_v [2][2 * DD];

  const int w = threadIdx.x >> 5;
  const int lane = threadIdx.x & 31;
  const int b = blockIdx.x * 2 + w;
  if (b >= B) return;                      // wave-uniform: EXEC stays all-1s for WMMA

  _Float16* act = sh_act[w];
  float* Sqh = sh_qh[w]; float* Sc = sh_c[w]; float* Sm = sh_m[w];
  float* P = sh_p[w];  float* V = sh_v[w];

  const int n0 = lane & 15;
  const int hi = lane >> 4;
  const int mo = hi ? 8 : 0;

  // ---- build x [16,28] -> padded f16 A tile (K padded to 32) ----
  {
    const float* ob = obs + (size_t)b * OBSW;
    int j = n0; int c0 = hi * 16;
    _Float16* row = act + j * ASTR;
#pragma unroll
    for (int t = 0; t < 16; ++t) {
      int c = c0 + t;
      float v = 0.f;
      if (c < 10)      v = ob[c];
      else if (c < 25) v = ob[10 + j * 15 + (c - 10)];
      else if (c < 28) v = ob[250 + j * 3 + (c - 25)];
      row[c] = (_Float16)v;
    }
  }

  // ---- mlp_in: resid = x @ W_in + b_in  (4 WMMAs) ----
  v8f resid[4];
  {
    v16h a0 = load_afrag(act, 0, lane);
    const v16h* fb = (const v16h*)fragWin;
#pragma unroll
    for (int nt = 0; nt < 4; ++nt) {
      v8f acc = {};
      acc = wmma16(a0, fb[nt * 32 + lane], acc);
      float bv = b_in[nt * 16 + n0];
#pragma unroll
      for (int r = 0; r < 8; ++r) acc[r] += bv;
      resid[nt] = acc;
    }
  }

  auto scatter_act = [&](v8f* fr) {       // C-layout frags -> row-major f16 tile
#pragma unroll
    for (int nt = 0; nt < 4; ++nt)
#pragma unroll
      for (int r = 0; r < 8; ++r)
        act[(r + mo) * ASTR + nt * 16 + n0] = (_Float16)fr[nt][r];
  };
  scatter_act(resid);

  // ---- 3 attentive graph blocks ----
  for (int blk = 0; blk < NBLK; ++blk) {
    const v16h* FB = (const v16h*)fragWbig + blk * 24 * 32;
    const float* bb = bbig + blk * 192;

    v16h a0 = load_afrag(act, 0, lane);
    v16h a1 = load_afrag(act, 1, lane);
    // qh|c|m = act @ WBIG + bbig : 12 N-tiles x 2 K-tiles = 24 WMMAs
#pragma unroll
    for (int nt = 0; nt < 12; ++nt) {
      v8f acc = {};
      acc = wmma16(a0, FB[(nt * 2 + 0) * 32 + lane], acc);
      acc = wmma16(a1, FB[(nt * 2 + 1) * 32 + lane], acc);
      float bv = bb[nt * 16 + n0];
      float* dst = (nt < 4) ? Sqh : (nt < 8 ? Sc : Sm);
      int col = (nt & 3) * 16 + n0;
#pragma unroll
      for (int r = 0; r < 8; ++r) dst[(r + mo) * FSTR + col] = acc[r] + bv;
    }

    // logits[i][j] = Wlg . tanh(c[j]+qh[i]) ; softmax over j
    {
      const float* wlg = Wlg + blk * DD;
      float tinv = 1.0f / temp[blk];
      float bl = blg[blk];
      const float* qrow = Sqh + n0 * FSTR;
      float lg[8];
#pragma unroll
      for (int t = 0; t < 8; ++t) {
        int j = t * 2 + hi;
        const float* crow = Sc + j * FSTR;
        float s = 0.f;
        for (int d = 0; d < DD; ++d) s += wlg[d] * fast_tanh(crow[d] + qrow[d]);
        lg[t] = (s + bl) * tinv;
      }
      float mx = lg[0];
#pragma unroll
      for (int t = 1; t < 8; ++t) mx = fmaxf(mx, lg[t]);
      mx = fmaxf(mx, __shfl_xor(mx, 16, 32));
      float sum = 0.f;
#pragma unroll
      for (int t = 0; t < 8; ++t) { lg[t] = __expf(lg[t] - mx); sum += lg[t]; }
      sum += __shfl_xor(sum, 16, 32);
      float inv = 1.0f / sum;
#pragma unroll
      for (int t = 0; t < 8; ++t) P[n0 * 17 + t * 2 + hi] = lg[t] * inv;
    }

    // heads = leaky(P @ m) -> f16 tile
    {
      const float* prow = P + n0 * 17;
      for (int dd = 0; dd < 32; ++dd) {
        int d = hi * 32 + dd;
        float s = 0.f;
#pragma unroll
        for (int j = 0; j < 16; ++j) s += prow[j] * Sm[j * FSTR + d];
        act[n0 * ASTR + d] = (_Float16)lrelu(s);
      }
    }

    // new = heads @ Wrh + brh ; resid = leaky(resid + new) -> next tile
    {
      v16h h0 = load_afrag(act, 0, lane);
      v16h h1 = load_afrag(act, 1, lane);
      const v16h* FR = (const v16h*)fragWrh + blk * 8 * 32;
      const float* br = brh + blk * DD;
#pragma unroll
      for (int nt = 0; nt < 4; ++nt) {
        v8f acc = {};
        acc = wmma16(h0, FR[(nt * 2 + 0) * 32 + lane], acc);
        acc = wmma16(h1, FR[(nt * 2 + 1) * 32 + lane], acc);
        float bv = br[nt * 16 + n0];
#pragma unroll
        for (int r = 0; r < 8; ++r) resid[nt][r] = lrelu(resid[nt][r] + acc[r] + bv);
      }
      scatter_act(resid);
    }
  }

  // ---- attentive pooling with constant query (qhp precomputed) ----
  {
#pragma unroll
    for (int nt = 0; nt < 4; ++nt)            // final out rows (f32) -> Sc
#pragma unroll
      for (int r = 0; r < 8; ++r)
        Sc[(r + mo) * FSTR + nt * 16 + n0] = resid[nt][r];

    const float* crow = Sc + n0 * FSTR;
    float s = 0.f;
    for (int dd = 0; dd < 32; ++dd) {
      int d = hi * 32 + dd;
      s += pWlg[d] * fast_tanh(crow[d] + qhp[d]);
    }
    s += __shfl_xor(s, 16, 32);
    float lgv = (s + pblg[0]) / ptemp[0];
    float mx = lgv;
#pragma unroll
    for (int off = 1; off < 16; off <<= 1) mx = fmaxf(mx, __shfl_xor(mx, off, 32));
    float e = __expf(lgv - mx);
    float sum = e;
#pragma unroll
    for (int off = 1; off < 16; off <<= 1) sum += __shfl_xor(sum, off, 32);
    P[n0] = e / sum;                           // lanes 16..31 duplicate-write same value

    // pooled_pre = leaky(sum_j p[j]*out[j][:])
#pragma unroll
    for (int t = 0; t < 2; ++t) {
      int d = lane + t * 32;
      float acc = 0.f;
#pragma unroll
      for (int j = 0; j < 16; ++j) acc += P[j] * Sc[j * FSTR + d];
      V[d] = lrelu(acc);
    }
  }

  // ---- dense tail: pWrh (no act), Wo1..Wo3 (relu), Wo4 -> 4 outputs ----
  auto fc = [&](const float* src, const float* W, const float* bias, float* dst, bool dorelu) {
#pragma unroll
    for (int t = 0; t < 2; ++t) {
      int n = lane + t * 32;
      float s = bias[n];
      for (int d = 0; d < DD; ++d) s += src[d] * W[d * DD + n];
      dst[n] = dorelu ? fmaxf(s, 0.f) : s;
    }
  };
  fc(V, pWrh, pbrh, V + DD, false);   // pooled
  fc(V + DD, Wo1, bo1, V, true);      // h1
  fc(V, Wo2, bo2, V + DD, true);      // h2
  fc(V + DD, Wo3, bo3, V, true);      // h3

  if (lane < 4) {
    float s = bo4[lane];
    for (int d = 0; d < DD; ++d) s += V[d] * Wo4[d * 4 + lane];
    out[(size_t)b * 4 + lane] = s;
  }
}

// ---------------- host launcher ----------------

extern "C" void kernel_launch(void* const* d_in, const int* in_sizes, int n_in,
                              void* d_out, int out_size, void* d_ws, size_t ws_size,
                              hipStream_t stream) {
  const float* obs  = (const float*)d_in[0];
  const float* W_in = (const float*)d_in[1];
  const float* b_in = (const float*)d_in[2];
  const float* Wqcm = (const float*)d_in[3];
  const float* bqcm = (const float*)d_in[4];
  const float* Wch  = (const float*)d_in[5];
  const float* bch  = (const float*)d_in[6];
  const float* Wlg  = (const float*)d_in[7];
  const float* blg  = (const float*)d_in[8];
  const float* Wrh  = (const float*)d_in[9];
  const float* brh  = (const float*)d_in[10];
  const float* temp = (const float*)d_in[11];
  const float* pq   = (const float*)d_in[12];
  const float* pWch = (const float*)d_in[13];
  const float* pbch = (const float*)d_in[14];
  const float* pWlg = (const float*)d_in[15];
  const float* pblg = (const float*)d_in[16];
  const float* pWrh = (const float*)d_in[17];
  const float* pbrh = (const float*)d_in[18];
  const float* ptemp= (const float*)d_in[19];
  const float* Wo1  = (const float*)d_in[20];
  const float* bo1  = (const float*)d_in[21];
  const float* Wo2  = (const float*)d_in[22];
  const float* bo2  = (const float*)d_in[23];
  const float* Wo3  = (const float*)d_in[24];
  const float* bo3  = (const float*)d_in[25];
  const float* Wo4  = (const float*)d_in[26];
  const float* bo4  = (const float*)d_in[27];

  const int B = in_sizes[0] / OBSW;

  // workspace layout (256B-aligned regions)
  char* ws = (char*)d_ws;
  float*     WBIG  = (float*)(ws);                 // 3*64*192*4 = 147456
  float*     BBIG  = (float*)(ws + 147456);        // 3*192*4   = 2304
  float*     QHP   = (float*)(ws + 149760);        // 64*4      = 256
  _Float16*  FWIN  = (_Float16*)(ws + 150016);     // 4*512*2   = 4096
  _Float16*  FWBIG = (_Float16*)(ws + 154112);     // 3*24*512*2= 73728
  _Float16*  FWRH  = (_Float16*)(ws + 227840);     // 3*8*512*2 = 24576

  prep_fuse<<<3 * 64, 192, 0, stream>>>(Wqcm, Wch, WBIG);
  prep_bias<<<4, 192, 0, stream>>>(bqcm, Wch, bch, pq, pWch, pbch, BBIG, QHP);
  pack_frags<<<1, 128, 0, stream>>>(W_in, 28, 64, 4, 1, FWIN);
  for (int i = 0; i < 3; ++i) {
    pack_frags<<<6, 128, 0, stream>>>(WBIG + (size_t)i * 64 * 192, 64, 192, 12, 2,
                                      FWBIG + (size_t)i * 24 * 512);
    pack_frags<<<2, 128, 0, stream>>>(Wrh + (size_t)i * 4096, 64, 64, 4, 2,
                                      FWRH + (size_t)i * 8 * 512);
  }

  actor_main<<<(B + 1) / 2, 64, 0, stream>>>(
      obs, b_in, Wlg, blg, temp, brh,
      pWlg, pblg, pWrh, pbrh, ptemp,
      Wo1, bo1, Wo2, bo2, Wo3, bo3, Wo4, bo4,
      FWIN, FWBIG, FWRH, BBIG, QHP,
      (float*)d_out, B);
}
